// MultiHeadAttention_38972533244088
// MI455X (gfx1250) — matrix operand order repaired and checked by reading, not compiled, not run on hardware
//
#include <hip/hip_runtime.h>
#include <hip/hip_bf16.h>

typedef __attribute__((ext_vector_type(16))) _Float16 v16h;
typedef __attribute__((ext_vector_type(8)))  float    v8f;

// Problem constants (match reference)
constexpr int BB  = 2;
constexpr int S   = 2048;
constexpr int D   = 1024;
constexpr int H   = 16;
constexpr int HD  = 64;          // head dim
constexpr int BS  = BB * S;      // 4096 rows
constexpr float SCALE = 0.125f;  // 1/sqrt(64)

// LDS B-tile row stride (halfs): 72 keeps b128 stores 16B-aligned (144B rows)
// and spreads fragment-read bank usage.
constexpr int BSTRIDE = 72;

// ---------------------------------------------------------------------------
// Fragment loaders (contiguous k order k = 16*hi + s for A and B, wave32)
// ---------------------------------------------------------------------------

// 16-bit A-matrix 16x32: lane r = lane&15 holds row M=r; hi = lane>>4 selects
// K-halves. VGPR j holds a K pair; as dwords: dj = j + 8*hi.
__device__ inline v16h load_a_frag(const _Float16* rowp, int hi) {
  v16h a;
  uint32_t* ap = (uint32_t*)&a;
  const uint32_t* src = (const uint32_t*)rowp;
#pragma unroll
  for (int j = 0; j < 8; ++j) {
    int dj = j + 8 * hi;
    ap[j] = src[dj];
  }
  return a;
}

// 16-bit B-matrix 32x16: lane c = lane&15 holds column N=c; lanes 0-15 hold
// K=0..15 in V0..V7 (2/VGPR), lanes 16-31 hold K=16..31.
// Contiguous case (B = Mrow^T): colp = &M[col, d0].
__device__ inline v16h load_b_frag_contig(const _Float16* colp, int hi) {
  v16h b;
  uint32_t* bp = (uint32_t*)&b;
  const uint32_t* src = (const uint32_t*)colp;
#pragma unroll
  for (int j = 0; j < 8; ++j) bp[j] = src[8 * hi + j];
  return b;
}

// Strided case: B row-major KxN, basep = &B[k0, n0+c], leading dim ldb (halfs).
__device__ inline v16h load_b_frag_strided(const _Float16* basep, int ldb, int hi) {
  v16h b;
#pragma unroll
  for (int j = 0; j < 8; ++j) {
    int k = hi * 16 + 2 * j;
    b[2 * j]     = basep[(size_t)k * ldb];
    b[2 * j + 1] = basep[(size_t)(k + 1) * ldb];
  }
  return b;
}

// B fragment from the LDS-staged tile (row-major, stride BSTRIDE halfs).
__device__ inline v16h load_b_frag_lds(const _Float16* basep, int hi) {
  v16h b;
#pragma unroll
  for (int j = 0; j < 8; ++j) {
    int k = hi * 16 + 2 * j;
    b[2 * j]     = basep[k * BSTRIDE];
    b[2 * j + 1] = basep[(k + 1) * BSTRIDE];
  }
  return b;
}

// ---------------------------------------------------------------------------
// Async global -> LDS copy (CDNA5, ASYNCcnt). Per ISA 10.2/15.18: the LDS
// destination is the low 32 bits of the flat address; the global address is a
// 64-bit VGPR pair. Tracked by ASYNCcnt, drained with s_wait_asynccnt.
// ---------------------------------------------------------------------------
__device__ inline void async_copy_b128(const _Float16* g, _Float16* l) {
  uint64_t ga = (uint64_t)(uintptr_t)g;
  uint32_t la = (uint32_t)(uintptr_t)l;   // LDS byte offset (flat addr[31:0])
  asm volatile("global_load_async_to_lds_b128 %0, %1, off"
               :: "v"(la), "v"(ga) : "memory");
}

__device__ inline void wait_async0() {
#if __has_builtin(__builtin_amdgcn_s_wait_asynccnt)
  __builtin_amdgcn_s_wait_asynccnt(0);
#else
  asm volatile("s_wait_asynccnt 0x0" ::: "memory");
#endif
}

// ---------------------------------------------------------------------------
// f32 -> f16 conversion
// ---------------------------------------------------------------------------
__global__ void f32_to_f16_kernel(const float* __restrict__ in,
                                  _Float16* __restrict__ out, int n) {
  int i = blockIdx.x * blockDim.x + threadIdx.x;
  if (i < n) out[i] = (_Float16)in[i];
}

// ---------------------------------------------------------------------------
// f16 WMMA GEMM: C[M,N] = A[M,K] @ B[K,N] (row-major, f32 accumulate).
// Block = 4 waves (128 threads). Each wave owns a 64x64 tile (16 acc frags,
// 16 WMMAs per 32-deep k-chunk); block covers 256x64.
// B chunks (32x64) are streamed into double-buffered LDS with async-to-LDS
// loads shared by all 4 waves.
// F32OUT: store f32 + bias (final projection); else store f16.
// ---------------------------------------------------------------------------
template <bool F32OUT>
__global__ void __launch_bounds__(128, 1)
gemm16_wmma(const _Float16* __restrict__ A,
            const _Float16* __restrict__ Bm,
            _Float16* __restrict__ Ch,
            float* __restrict__ Cf,
            const float* __restrict__ bias,
            int M, int N, int K) {
  __shared__ _Float16 Bs[2][32 * BSTRIDE];  // 2 x 4.5 KB

  const int tid  = threadIdx.x;
  const int lane = tid & 31;
  const int wave = tid >> 5;
  const int m0 = (blockIdx.y * 4 + wave) * 64;
  const int n0 = blockIdx.x * 64;
  const int r  = lane & 15;
  const int hi = lane >> 4;

  // staging slice for this thread: row sk (0..31), 16-half group sg (0..3)
  const int sk = tid >> 2;
  const int sg = tid & 3;

  v8f acc[4][4] = {};  // [row sub-tile s][col sub-tile t]

  const _Float16* arow[4];
#pragma unroll
  for (int s = 0; s < 4; ++s)
    arow[s] = A + (size_t)(m0 + 16 * s + r) * K;

  const int nch = K / 32;

  // prefetch chunk 0
  {
    const _Float16* g = Bm + (size_t)sk * N + n0 + 16 * sg;
    _Float16* l = &Bs[0][sk * BSTRIDE + 16 * sg];
    async_copy_b128(g, l);
    async_copy_b128(g + 8, l + 8);
  }

  for (int i = 0; i < nch; ++i) {
    const int buf = i & 1;
    wait_async0();       // chunk i landed in Bs[buf]
    __syncthreads();     // ...for every wave in the block

    if (i + 1 < nch) {   // stream chunk i+1 into the other buffer
      const _Float16* g = Bm + (size_t)((i + 1) * 32 + sk) * N + n0 + 16 * sg;
      _Float16* l = &Bs[buf ^ 1][sk * BSTRIDE + 16 * sg];
      async_copy_b128(g, l);
      async_copy_b128(g + 8, l + 8);
    }

    const int k0 = i * 32;
    v16h a[4];
#pragma unroll
    for (int s = 0; s < 4; ++s) a[s] = load_a_frag(arow[s] + k0, hi);

#pragma unroll
    for (int t = 0; t < 4; ++t) {
      v16h b = load_b_frag_lds(&Bs[buf][16 * t + r], hi);
#pragma unroll
      for (int s = 0; s < 4; ++s) {
        acc[s][t] = __builtin_amdgcn_wmma_f32_16x16x32_f16(
            false, a[s], false, b, (short)0, acc[s][t], false, false);
      }
    }
    __syncthreads();     // all waves done with Bs[buf] before it is refilled
  }

  // C/D layout: VGPR j holds row (+ j + 8*hi), column n = lane&15 per 16-tile.
#pragma unroll
  for (int s = 0; s < 4; ++s) {
#pragma unroll
    for (int t = 0; t < 4; ++t) {
#pragma unroll
      for (int j = 0; j < 8; ++j) {
        int m = m0 + 16 * s + j + 8 * hi;
        int n = n0 + 16 * t + r;
        if constexpr (F32OUT) {
          Cf[(size_t)m * N + n] = acc[s][t][j] + bias[n];
        } else {
          Ch[(size_t)m * N + n] = (_Float16)acc[s][t][j];
        }
      }
    }
  }
}

// ---------------------------------------------------------------------------
// Causal flash attention, head_dim = 64, 32 q-rows per wave (one wave/block).
// Grid: (S/32, H, B). Q/K/V/Ctx are f16 [B, S, D], head h at column h*HD.
//
// Register-pressure design: score tiles go to LDS as soon as they are
// produced; the per-row softmax is done in TWO passes over the LDS row
// (pass 1: masked max; pass 2: exp+sum with f16 pairs packed straight into
// the P buffer), so no 32-element register arrays exist. K fragments are
// loaded per kv sub-tile, not all at once. Each of the 32 lanes owns one
// q-row -- no cross-lane shuffles at all.
// ---------------------------------------------------------------------------
__global__ void __launch_bounds__(32, 1)
attn64_wmma(const _Float16* __restrict__ Q,
            const _Float16* __restrict__ Km,
            const _Float16* __restrict__ V,
            _Float16* __restrict__ Ctx) {
  __shared__ float    Sbuf[32 * 36];  // scores, padded stride (bank spread)
  __shared__ uint32_t Pbuf[32 * 18];  // packed f16 probability pairs
  __shared__ float    Abuf[32];       // per-row alpha broadcast
  __shared__ float    Lbuf[32];       // per-row l broadcast (epilogue)

  const int lane = threadIdx.x & 31;
  const int r  = lane & 15;
  const int hi = lane >> 4;
  const int q0 = blockIdx.x * 32;
  const int h  = blockIdx.y;
  const int bb = blockIdx.z;

  const size_t base = (size_t)bb * S * D + (size_t)h * HD;

  // Q A-fragments: 2 row sub-tiles x 2 d-halves, loaded once.
  v16h qa[2][2];
#pragma unroll
  for (int s = 0; s < 2; ++s) {
    const _Float16* qrow = Q + base + (size_t)(q0 + 16 * s + r) * D;
    qa[s][0] = load_a_frag(qrow, hi);
    qa[s][1] = load_a_frag(qrow + 32, hi);
  }

  v8f acc[2][4] = {};          // [row sub-tile][d sub-tile]
  float m_l = -1e30f;          // stats of the row this lane owns
  float l_l = 0.0f;
  const int myrow = q0 + lane;

  const int nch = q0 / 32 + 1;  // causal: chunks with c0 <= q0+31
  for (int ci = 0; ci < nch; ++ci) {
    const int c0 = ci * 32;

    // --- scores: per kv sub-tile load K frags, 2 WMMAs per row sub-tile ---
#pragma unroll
    for (int t = 0; t < 2; ++t) {
      const _Float16* krow = Km + base + (size_t)(c0 + 16 * t + r) * D;
      v16h kb0 = load_b_frag_contig(krow, hi);       // d 0..31
      v16h kb1 = load_b_frag_contig(krow + 32, hi);  // d 32..63
#pragma unroll
      for (int s = 0; s < 2; ++s) {
        v8f st = {};
        st = __builtin_amdgcn_wmma_f32_16x16x32_f16(
            false, qa[s][0], false, kb0, (short)0, st, false, false);
        st = __builtin_amdgcn_wmma_f32_16x16x32_f16(
            false, qa[s][1], false, kb1, (short)0, st, false, false);
#pragma unroll
        for (int j = 0; j < 8; ++j)
          Sbuf[(16 * s + j + 8 * hi) * 36 + 16 * t + r] = st[j];
      }
    }
    __syncthreads();

    // --- row-serial softmax: this lane owns q-row 'myrow' ---
    const float4* srow4 = (const float4*)&Sbuf[lane * 36];
    const int lim = myrow - c0;    // kv index k valid iff k <= lim (lim >= 0)

    // pass 1: masked max (tree per float4, no register arrays)
    float mx = -1e30f;
#pragma unroll
    for (int e = 0; e < 8; ++e) {
      float4 f = srow4[e];
      float v0 = ((4 * e + 0) <= lim) ? f.x * SCALE : -1e30f;
      float v1 = ((4 * e + 1) <= lim) ? f.y * SCALE : -1e30f;
      float v2 = ((4 * e + 2) <= lim) ? f.z * SCALE : -1e30f;
      float v3 = ((4 * e + 3) <= lim) ? f.w * SCALE : -1e30f;
      mx = fmaxf(mx, fmaxf(fmaxf(v0, v1), fmaxf(v2, v3)));
    }
    const float mnew = fmaxf(m_l, mx);
    const float al   = __expf(m_l - mnew);

    // pass 2: exp + sum, pack f16 pairs straight into Pbuf
    uint32_t* prow = &Pbuf[lane * 18];
    float ps = 0.0f;
#pragma unroll
    for (int e = 0; e < 8; ++e) {
      float4 f = srow4[e];
      float p0 = __expf((((4 * e + 0) <= lim) ? f.x * SCALE : -1e30f) - mnew);
      float p1 = __expf((((4 * e + 1) <= lim) ? f.y * SCALE : -1e30f) - mnew);
      float p2 = __expf((((4 * e + 2) <= lim) ? f.z * SCALE : -1e30f) - mnew);
      float p3 = __expf((((4 * e + 3) <= lim) ? f.w * SCALE : -1e30f) - mnew);
      ps += (p0 + p1) + (p2 + p3);
      union { _Float16 hh[2]; uint32_t u; } pka, pkb;
      pka.hh[0] = (_Float16)p0; pka.hh[1] = (_Float16)p1;
      pkb.hh[0] = (_Float16)p2; pkb.hh[1] = (_Float16)p3;
      prow[2 * e]     = pka.u;
      prow[2 * e + 1] = pkb.u;
    }
    l_l = l_l * al + ps;
    m_l = mnew;
    Abuf[lane] = al;
    __syncthreads();

    // --- reassemble P A-fragments (row-per-lane; dwords {8hi..8hi+7}) ---
    v16h pa[2];
#pragma unroll
    for (int s = 0; s < 2; ++s) {
      const uint32_t* pr = &Pbuf[(16 * s + r) * 18];
      uint32_t* pd = (uint32_t*)&pa[s];
#pragma unroll
      for (int j = 0; j < 4; ++j) pd[j] = pr[8 * hi + j];
#pragma unroll
      for (int j = 0; j < 4; ++j) pd[4 + j] = pr[8 * hi + 4 + j];
    }

    // --- alpha back to C-layout (rows j + 8*hi of each sub-tile) ---
    float alphav[2][8];
#pragma unroll
    for (int s = 0; s < 2; ++s) {
      const float4* ap = (const float4*)&Abuf[16 * s + 8 * hi];
      float4 a0 = ap[0], a1 = ap[1];
      alphav[s][0] = a0.x; alphav[s][1] = a0.y; alphav[s][2] = a0.z; alphav[s][3] = a0.w;
      alphav[s][4] = a1.x; alphav[s][5] = a1.y; alphav[s][6] = a1.z; alphav[s][7] = a1.w;
    }

    // --- PV: acc[s][t] = alpha*acc[s][t] + P(16x32) @ V_chunk(32x16) ---
#pragma unroll
    for (int t = 0; t < 4; ++t) {
      v16h vb = load_b_frag_strided(V + base + (size_t)c0 * D + (16 * t + r), D, hi);
#pragma unroll
      for (int s = 0; s < 2; ++s) {
        v8f at;
#pragma unroll
        for (int j = 0; j < 8; ++j) at[j] = acc[s][t][j] * alphav[s][j];
        acc[s][t] = __builtin_amdgcn_wmma_f32_16x16x32_f16(
            false, pa[s], false, vb, (short)0, at, false, false);
      }
    }
    __syncthreads();  // Sbuf/Pbuf/Abuf reused next chunk
  }

  // --- epilogue: broadcast l, normalize, store ctx (f16) ---
  Lbuf[lane] = l_l;
  __syncthreads();
  float lv[2][8];
#pragma unroll
  for (int s = 0; s < 2; ++s) {
    const float4* lp = (const float4*)&Lbuf[16 * s + 8 * hi];
    float4 a0 = lp[0], a1 = lp[1];
    lv[s][0] = a0.x; lv[s][1] = a0.y; lv[s][2] = a0.z; lv[s][3] = a0.w;
    lv[s][4] = a1.x; lv[s][5] = a1.y; lv[s][6] = a1.z; lv[s][7] = a1.w;
  }
#pragma unroll
  for (int s = 0; s < 2; ++s) {
#pragma unroll
    for (int t = 0; t < 4; ++t) {
#pragma unroll
      for (int j = 0; j < 8; ++j) {
        float o = acc[s][t][j] / lv[s][j];
        Ctx[base + (size_t)(q0 + 16 * s + j + 8 * hi) * D + 16 * t + r] = (_Float16)o;
      }
    }
  }
}

// ---------------------------------------------------------------------------
// Launch
// ---------------------------------------------------------------------------
extern "C" void kernel_launch(void* const* d_in, const int* in_sizes, int n_in,
                              void* d_out, int out_size, void* d_ws, size_t ws_size,
                              hipStream_t stream) {
  (void)in_sizes; (void)n_in; (void)out_size; (void)ws_size;

  const float* x  = (const float*)d_in[0];
  const float* Wq = (const float*)d_in[1];
  const float* Wk = (const float*)d_in[2];
  const float* Wv = (const float*)d_in[3];
  const float* Wo = (const float*)d_in[4];
  const float* bo = (const float*)d_in[5];
  float* out = (float*)d_out;

  // Workspace layout (halfs): xh | Wq | Wk | Wv | Wo | Q | K | V | ctx (~48MB)
  _Float16* ws   = (_Float16*)d_ws;
  _Float16* xh   = ws;
  _Float16* wqh  = xh  + (size_t)BS * D;
  _Float16* wkh  = wqh + (size_t)D * D;
  _Float16* wvh  = wkh + (size_t)D * D;
  _Float16* woh  = wvh + (size_t)D * D;
  _Float16* qh   = woh + (size_t)D * D;
  _Float16* kh   = qh  + (size_t)BS * D;
  _Float16* vh   = kh  + (size_t)BS * D;
  _Float16* ctxh = vh  + (size_t)BS * D;

  // 1) convert inputs to f16
  {
    int n = BS * D;
    f32_to_f16_kernel<<<(n + 255) / 256, 256, 0, stream>>>(x, xh, n);
    n = D * D;
    f32_to_f16_kernel<<<(n + 255) / 256, 256, 0, stream>>>(Wq, wqh, n);
    f32_to_f16_kernel<<<(n + 255) / 256, 256, 0, stream>>>(Wk, wkh, n);
    f32_to_f16_kernel<<<(n + 255) / 256, 256, 0, stream>>>(Wv, wvh, n);
    f32_to_f16_kernel<<<(n + 255) / 256, 256, 0, stream>>>(Wo, woh, n);
  }

  // 2) Q/K/V projections: [4096,1024] = [4096,1024] @ [1024,1024]
  dim3 gemm_grid(D / 64, BS / 256);  // (16, 16)
  gemm16_wmma<false><<<gemm_grid, 128, 0, stream>>>(xh, wqh, qh, nullptr, nullptr, BS, D, D);
  gemm16_wmma<false><<<gemm_grid, 128, 0, stream>>>(xh, wkh, kh, nullptr, nullptr, BS, D, D);
  gemm16_wmma<false><<<gemm_grid, 128, 0, stream>>>(xh, wvh, vh, nullptr, nullptr, BS, D, D);

  // 3) causal flash attention per (b, h, 32-row q tile)
  dim3 attn_grid(S / 32, H, BB);     // (64, 16, 2)
  attn64_wmma<<<attn_grid, 32, 0, stream>>>(qh, kh, vh, ctxh);

  // 4) output projection with bias -> f32
  gemm16_wmma<true><<<gemm_grid, 128, 0, stream>>>(ctxh, woh, nullptr, out, bo, BS, D, D);
}
